// Net_17609365913696
// MI455X (gfx1250) — compile-verified, matrix operand order checked by reading, not run on hardware
//
#include <hip/hip_runtime.h>
#include <hip/hip_bf16.h>
#include <math.h>

// ---------------------------------------------------------------------------
// Problem constants (from the reference)
// ---------------------------------------------------------------------------
#define NN   50000      // nodes
#define EE   800000     // edges
#define ESL  850000     // edges + self loops
#define CC   256        // layer size
#define DD   2048       // input feature dim
#define NEG_SLOPE 0.2f

typedef __attribute__((ext_vector_type(16))) __bf16 v16bf;
typedef __attribute__((ext_vector_type(8)))  float  v8f;

struct F4 { float x, y, z, w; };            // POD 16-byte vector
union FragU { struct { F4 a, b; } f; v16bf v; };

static __device__ __forceinline__ v8f wmma_bf16(v16bf a, v16bf b, v8f c) {
    // D = A x B + C, 16x16x32, bf16 inputs, f32 accumulate
    return __builtin_amdgcn_wmma_f32_16x16x32_bf16(false, a, false, b,
                                                   (short)0, c, false, false);
}

// Load one wave32 WMMA fragment (A-style layout) from an LDS row base.
// lane<16 : K = 0..7 and 16..23 ; lane>=16 : K = 8..15 and 24..31
static __device__ __forceinline__ v16bf load_frag(const __bf16* rowBase, int lh) {
    FragU u;
    u.f.a = *(const F4*)(rowBase + lh * 8);
    u.f.b = *(const F4*)(rowBase + 16 + lh * 8);
    return u.v;
}

static __device__ __forceinline__ float leaky(float v) {
    return v > 0.0f ? v : NEG_SLOPE * v;
}

// monotonic float<->uint encoding for atomicMax on floats
static __device__ __forceinline__ unsigned fenc(float f) {
    unsigned u = __float_as_uint(f);
    return (u & 0x80000000u) ? ~u : (u | 0x80000000u);
}
static __device__ __forceinline__ float fdec(unsigned u) {
    return (u & 0x80000000u) ? __uint_as_float(u & 0x7fffffffu)
                             : __uint_as_float(~u);
}
#define ENC_NEG_INF 0x007FFFFFu   // fenc(-inf)

// ---------------------------------------------------------------------------
// Generic WMMA GEMM:  C[M x 256] = A[M x K] @ B[K x 256] + bias
// block = 256 threads (8 waves), tile 128x64, BK = 32, software pipelined.
// Branch-free A-row handling: OOB rows clamp to M-1 (their outputs are
// never stored, so the values are irrelevant).
// ---------------------------------------------------------------------------
__global__ void gemm_bias_kernel(const float* __restrict__ A,
                                 const float* __restrict__ B,
                                 const float* __restrict__ bias,
                                 float* __restrict__ C,
                                 int M, int K) {
    __shared__ alignas(16) __bf16 As[128 * 32];
    __shared__ alignas(16) __bf16 Bs[64 * 32];   // transposed: Bs[n][k]

    const int t    = threadIdx.x;                // 0..255
    const int wave = t >> 5, lane = t & 31;
    const int wm   = wave >> 1, wn = wave & 1;   // 4 x 2 wave grid
    const int l16  = lane & 15, lh = lane >> 4;
    const int rowBase = blockIdx.x * 128;
    const int colBase = blockIdx.y * 64;

    v8f acc[2][2] = {};
    F4 ra[4], rb[2];

    // ---- register-stage loaders (branch free) ----
    auto loadA = [&](int kb) {
#pragma unroll
        for (int i = 0; i < 4; ++i) {
            int e4 = t + 256 * i;                // 0..1023 float4 groups
            int r  = e4 >> 3;                    // 0..127
            int k4 = (e4 & 7) << 2;
            int gr = min(rowBase + r, M - 1);    // clamp, no branch
            ra[i] = *(const F4*)(A + (size_t)gr * K + kb + k4);
        }
    };
    auto loadB = [&](int kb) {
#pragma unroll
        for (int i = 0; i < 2; ++i) {
            int e4 = t + 256 * i;                // 0..511
            int k  = e4 >> 4;
            int n4 = (e4 & 15) << 2;
            rb[i] = *(const F4*)(B + (size_t)(kb + k) * CC + colBase + n4);
        }
    };
    auto stageLDS = [&]() {
#pragma unroll
        for (int i = 0; i < 4; ++i) {
            int e4 = t + 256 * i;
            int r  = e4 >> 3;
            int k4 = (e4 & 7) << 2;
            __bf16* p = &As[r * 32 + k4];
            p[0] = (__bf16)ra[i].x; p[1] = (__bf16)ra[i].y;
            p[2] = (__bf16)ra[i].z; p[3] = (__bf16)ra[i].w;
        }
#pragma unroll
        for (int i = 0; i < 2; ++i) {
            int e4 = t + 256 * i;
            int k  = e4 >> 4;
            int n4 = (e4 & 15) << 2;
            Bs[(n4 + 0) * 32 + k] = (__bf16)rb[i].x;
            Bs[(n4 + 1) * 32 + k] = (__bf16)rb[i].y;
            Bs[(n4 + 2) * 32 + k] = (__bf16)rb[i].z;
            Bs[(n4 + 3) * 32 + k] = (__bf16)rb[i].w;
        }
    };

    loadA(0); loadB(0);
    for (int kb = 0; kb < K; kb += 32) {
        stageLDS();
        __syncthreads();
        if (kb + 32 < K) { loadA(kb + 32); loadB(kb + 32); }  // overlap w/ WMMA

        v16bf a0 = load_frag(&As[(wm * 32 +      l16) * 32], lh);
        v16bf a1 = load_frag(&As[(wm * 32 + 16 + l16) * 32], lh);
        v16bf b0 = load_frag(&Bs[(wn * 32 +      l16) * 32], lh);
        v16bf b1 = load_frag(&Bs[(wn * 32 + 16 + l16) * 32], lh);

        acc[0][0] = wmma_bf16(a0, b0, acc[0][0]);
        acc[0][1] = wmma_bf16(a0, b1, acc[0][1]);
        acc[1][0] = wmma_bf16(a1, b0, acc[1][0]);
        acc[1][1] = wmma_bf16(a1, b1, acc[1][1]);
        __syncthreads();
    }

    // ---- epilogue: bias + store ----
#pragma unroll
    for (int im = 0; im < 2; ++im)
#pragma unroll
        for (int in = 0; in < 2; ++in)
#pragma unroll
            for (int r = 0; r < 8; ++r) {
                int row = rowBase + wm * 32 + im * 16 + r + 8 * lh;
                int col = colBase + wn * 32 + in * 16 + l16;
                if (row < M)
                    C[(size_t)row * CC + col] = acc[im][in][r] + bias[col];
            }
}

// ---------------------------------------------------------------------------
// Edge predictor GEMM (fused): rows are gathered concat(h[src],h[dst]) (K=512),
// B = Wp1 [512 x 256].  Epilogue: relu(z + bp1) . wp2 reduced per edge and
// atomically accumulated into out[e] (pre-initialized to bp2).
// E = 800000 is divisible by 128: no row guards anywhere.
// ---------------------------------------------------------------------------
__global__ void pred_gemm_kernel(const float* __restrict__ H,
                                 const int* __restrict__ src,
                                 const int* __restrict__ dst,
                                 const float* __restrict__ Wp1,
                                 const float* __restrict__ bp1,
                                 const float* __restrict__ wp2,
                                 float* __restrict__ out) {
    __shared__ alignas(16) __bf16 As[128 * 32];
    __shared__ alignas(16) __bf16 Bs[64 * 32];

    const int t    = threadIdx.x;
    const int wave = t >> 5, lane = t & 31;
    const int wm   = wave >> 1, wn = wave & 1;
    const int l16  = lane & 15, lh = lane >> 4;
    const int edgeBase = blockIdx.x * 128;
    const int colBase  = blockIdx.y * 64;

    v8f acc[2][2] = {};
    F4 ra[4], rb[2];

    auto loadA = [&](int kb) {
        const int* sel  = (kb < 256) ? src : dst;
        const int  koff = kb & 255;
#pragma unroll
        for (int i = 0; i < 4; ++i) {
            int e4 = t + 256 * i;
            int r  = e4 >> 3;
            int k4 = (e4 & 7) << 2;
            int node = sel[edgeBase + r];
            ra[i] = *(const F4*)(H + (size_t)node * CC + koff + k4);
        }
    };
    auto loadB = [&](int kb) {
#pragma unroll
        for (int i = 0; i < 2; ++i) {
            int e4 = t + 256 * i;
            int k  = e4 >> 4;
            int n4 = (e4 & 15) << 2;
            rb[i] = *(const F4*)(Wp1 + (size_t)(kb + k) * CC + colBase + n4);
        }
    };
    auto stageLDS = [&]() {
#pragma unroll
        for (int i = 0; i < 4; ++i) {
            int e4 = t + 256 * i;
            int r  = e4 >> 3;
            int k4 = (e4 & 7) << 2;
            __bf16* p = &As[r * 32 + k4];
            p[0] = (__bf16)ra[i].x; p[1] = (__bf16)ra[i].y;
            p[2] = (__bf16)ra[i].z; p[3] = (__bf16)ra[i].w;
        }
#pragma unroll
        for (int i = 0; i < 2; ++i) {
            int e4 = t + 256 * i;
            int k  = e4 >> 4;
            int n4 = (e4 & 15) << 2;
            Bs[(n4 + 0) * 32 + k] = (__bf16)rb[i].x;
            Bs[(n4 + 1) * 32 + k] = (__bf16)rb[i].y;
            Bs[(n4 + 2) * 32 + k] = (__bf16)rb[i].z;
            Bs[(n4 + 3) * 32 + k] = (__bf16)rb[i].w;
        }
    };

    loadA(0); loadB(0);
    for (int kb = 0; kb < 512; kb += 32) {
        stageLDS();
        __syncthreads();
        if (kb + 32 < 512) { loadA(kb + 32); loadB(kb + 32); }

        v16bf a0 = load_frag(&As[(wm * 32 +      l16) * 32], lh);
        v16bf a1 = load_frag(&As[(wm * 32 + 16 + l16) * 32], lh);
        v16bf b0 = load_frag(&Bs[(wn * 32 +      l16) * 32], lh);
        v16bf b1 = load_frag(&Bs[(wn * 32 + 16 + l16) * 32], lh);

        acc[0][0] = wmma_bf16(a0, b0, acc[0][0]);
        acc[0][1] = wmma_bf16(a0, b1, acc[0][1]);
        acc[1][0] = wmma_bf16(a1, b0, acc[1][0]);
        acc[1][1] = wmma_bf16(a1, b1, acc[1][1]);
        __syncthreads();
    }

    // epilogue: relu(z + bp1) * wp2, reduce across the 16 column lanes
    float bcol[2], wcol[2];
#pragma unroll
    for (int in = 0; in < 2; ++in) {
        int col = colBase + wn * 32 + in * 16 + l16;
        bcol[in] = bp1[col];
        wcol[in] = wp2[col];
    }
#pragma unroll
    for (int im = 0; im < 2; ++im)
#pragma unroll
        for (int r = 0; r < 8; ++r) {
            float s = 0.0f;
#pragma unroll
            for (int in = 0; in < 2; ++in) {
                float z = acc[im][in][r] + bcol[in];
                z = z > 0.0f ? z : 0.0f;
                s += z * wcol[in];
            }
            s += __shfl_xor(s, 8, 32);
            s += __shfl_xor(s, 4, 32);
            s += __shfl_xor(s, 2, 32);
            s += __shfl_xor(s, 1, 32);
            if (l16 == 0) {
                int e = edgeBase + wm * 32 + im * 16 + r + 8 * lh;
                atomicAdd(&out[e], s);
            }
        }
}

// ---------------------------------------------------------------------------
// Attention (scatter/gather, bandwidth bound) kernels
// ---------------------------------------------------------------------------
__global__ void node_scores_kernel(const float* __restrict__ h,
                                   const float* __restrict__ a_src,
                                   const float* __restrict__ a_dst,
                                   float* __restrict__ ssrc,
                                   float* __restrict__ sdst, int n) {
    int gid  = blockIdx.x * blockDim.x + threadIdx.x;
    int node = gid >> 5;
    int lane = threadIdx.x & 31;
    if (node >= n) return;
    const float* hp = h + (size_t)node * CC + lane * 8;
    F4 h0 = *(const F4*)hp, h1 = *(const F4*)(hp + 4);
    F4 s0 = *(const F4*)(a_src + lane * 8), s1 = *(const F4*)(a_src + lane * 8 + 4);
    F4 d0 = *(const F4*)(a_dst + lane * 8), d1 = *(const F4*)(a_dst + lane * 8 + 4);
    float vs = h0.x*s0.x + h0.y*s0.y + h0.z*s0.z + h0.w*s0.w
             + h1.x*s1.x + h1.y*s1.y + h1.z*s1.z + h1.w*s1.w;
    float vd = h0.x*d0.x + h0.y*d0.y + h0.z*d0.z + h0.w*d0.w
             + h1.x*d1.x + h1.y*d1.y + h1.z*d1.z + h1.w*d1.w;
#pragma unroll
    for (int m = 16; m >= 1; m >>= 1) {
        vs += __shfl_xor(vs, m, 32);
        vd += __shfl_xor(vd, m, 32);
    }
    if (lane == 0) { ssrc[node] = vs; sdst[node] = vd; }
}

static __device__ __forceinline__ void edge_sd(int idx, const int* src,
                                               const int* dst, int& s, int& d) {
    if (idx < EE) { s = src[idx]; d = dst[idx]; }
    else          { s = idx - EE; d = idx - EE; }   // self loops
}

__global__ void edge_max_kernel(const float* __restrict__ ssrc,
                                const float* __restrict__ sdst,
                                const int* __restrict__ src,
                                const int* __restrict__ dst,
                                unsigned* __restrict__ emax) {
    int idx = blockIdx.x * blockDim.x + threadIdx.x;
    if (idx >= ESL) return;
    int s, d; edge_sd(idx, src, dst, s, d);
    float v = leaky(ssrc[s] + sdst[d]);
    atomicMax(&emax[d], fenc(v));
}

__global__ void edge_exp_kernel(const float* __restrict__ ssrc,
                                const float* __restrict__ sdst,
                                const int* __restrict__ src,
                                const int* __restrict__ dst,
                                const unsigned* __restrict__ emax,
                                float* __restrict__ denom,
                                float* __restrict__ ex) {
    int idx = blockIdx.x * blockDim.x + threadIdx.x;
    if (idx >= ESL) return;
    int s, d; edge_sd(idx, src, dst, s, d);
    float v = leaky(ssrc[s] + sdst[d]);
    float e = __expf(v - fdec(emax[d]));
    ex[idx] = e;
    atomicAdd(&denom[d], e);
}

__global__ void edge_agg_kernel(const float* __restrict__ h,
                                const int* __restrict__ src,
                                const int* __restrict__ dst,
                                const float* __restrict__ ex,
                                const float* __restrict__ denom,
                                float* __restrict__ g) {
    int tid = blockIdx.x * blockDim.x + threadIdx.x;   // (edge, 4-channel chunk)
    int e   = tid >> 6;
    int c4  = (tid & 63) << 2;
    if (e >= ESL) return;
    int s, d; edge_sd(e, src, dst, s, d);
    float alpha = ex[e] / (denom[d] + 1e-16f);
    F4 hv = *(const F4*)(h + (size_t)s * CC + c4);
    float* gp = g + (size_t)d * CC + c4;
    atomicAdd(gp + 0, alpha * hv.x);
    atomicAdd(gp + 1, alpha * hv.y);
    atomicAdd(gp + 2, alpha * hv.z);
    atomicAdd(gp + 3, alpha * hv.w);
}

__global__ void add_bias_kernel(float* __restrict__ g,
                                const float* __restrict__ b, int n) {
    int i = blockIdx.x * blockDim.x + threadIdx.x;
    if (i < n) g[i] += b[i & (CC - 1)];
}

__global__ void fill_f32_kernel(float* p, float v, int n) {
    int i = blockIdx.x * blockDim.x + threadIdx.x;
    if (i < n) p[i] = v;
}
__global__ void fill_u32_kernel(unsigned* p, unsigned v, int n) {
    int i = blockIdx.x * blockDim.x + threadIdx.x;
    if (i < n) p[i] = v;
}
__global__ void pred_init_kernel(float* out, const float* bp2, int n) {
    int i = blockIdx.x * blockDim.x + threadIdx.x;
    if (i < n) out[i] = bp2[0];
}

// ---------------------------------------------------------------------------
// Host orchestration
// ---------------------------------------------------------------------------
extern "C" void kernel_launch(void* const* d_in, const int* in_sizes, int n_in,
                              void* d_out, int out_size, void* d_ws, size_t ws_size,
                              hipStream_t stream) {
    (void)in_sizes; (void)n_in; (void)out_size; (void)ws_size;

    const float* x      = (const float*)d_in[0];
    const int*   ei     = (const int*)d_in[1];
    const float* W1     = (const float*)d_in[2];
    const float* a_src1 = (const float*)d_in[3];
    const float* a_dst1 = (const float*)d_in[4];
    const float* b1     = (const float*)d_in[5];
    const float* W2     = (const float*)d_in[6];
    const float* a_src2 = (const float*)d_in[7];
    const float* a_dst2 = (const float*)d_in[8];
    const float* b2     = (const float*)d_in[9];
    const float* Wp1    = (const float*)d_in[10];
    const float* bp1    = (const float*)d_in[11];
    const float* wp2    = (const float*)d_in[12];
    const float* bp2    = (const float*)d_in[13];
    float*       out    = (float*)d_out;

    const int* src = ei;
    const int* dst = ei + EE;

    // workspace layout (floats)
    float*    ws    = (float*)d_ws;
    float*    bufA  = ws;                       // h (GEMM out), N*C
    float*    bufB  = ws + (size_t)NN * CC;     // aggregate,    N*C
    float*    ssrc  = ws + 2 * (size_t)NN * CC;
    float*    sdst  = ssrc + NN;
    unsigned* emax  = (unsigned*)(sdst + NN);
    float*    denom = (float*)(emax + NN);
    float*    ex    = denom + NN;               // ESL floats

    const dim3 gemmBlk(256);
    const dim3 gemmGrd((NN + 127) / 128, CC / 64);
    const int  TB = 256;
    const int  scoreBlocks = (NN * 32 + TB - 1) / TB;
    const int  edgeBlocks  = (ESL + TB - 1) / TB;
    const int  aggBlocks   = ((size_t)ESL * 64 + TB - 1) / TB;
    const int  ncBlocks    = (NN * CC + TB - 1) / TB;

    // ---------------- GAT layer 1 ----------------
    gemm_bias_kernel<<<gemmGrd, gemmBlk, 0, stream>>>(x, W1, b1, bufA, NN, DD);
    node_scores_kernel<<<scoreBlocks, TB, 0, stream>>>(bufA, a_src1, a_dst1, ssrc, sdst, NN);
    fill_u32_kernel<<<(NN + TB - 1) / TB, TB, 0, stream>>>(emax, ENC_NEG_INF, NN);
    fill_f32_kernel<<<(NN + TB - 1) / TB, TB, 0, stream>>>(denom, 0.0f, NN);
    fill_f32_kernel<<<ncBlocks, TB, 0, stream>>>(bufB, 0.0f, NN * CC);
    edge_max_kernel<<<edgeBlocks, TB, 0, stream>>>(ssrc, sdst, src, dst, emax);
    edge_exp_kernel<<<edgeBlocks, TB, 0, stream>>>(ssrc, sdst, src, dst, emax, denom, ex);
    edge_agg_kernel<<<aggBlocks, TB, 0, stream>>>(bufA, src, dst, ex, denom, bufB);
    add_bias_kernel<<<ncBlocks, TB, 0, stream>>>(bufB, b1, NN * CC);

    // ---------------- GAT layer 2 ----------------
    gemm_bias_kernel<<<gemmGrd, gemmBlk, 0, stream>>>(bufB, W2, b2, bufA, NN, CC);
    node_scores_kernel<<<scoreBlocks, TB, 0, stream>>>(bufA, a_src2, a_dst2, ssrc, sdst, NN);
    fill_u32_kernel<<<(NN + TB - 1) / TB, TB, 0, stream>>>(emax, ENC_NEG_INF, NN);
    fill_f32_kernel<<<(NN + TB - 1) / TB, TB, 0, stream>>>(denom, 0.0f, NN);
    fill_f32_kernel<<<ncBlocks, TB, 0, stream>>>(bufB, 0.0f, NN * CC);
    edge_max_kernel<<<edgeBlocks, TB, 0, stream>>>(ssrc, sdst, src, dst, emax);
    edge_exp_kernel<<<edgeBlocks, TB, 0, stream>>>(ssrc, sdst, src, dst, emax, denom, ex);
    edge_agg_kernel<<<aggBlocks, TB, 0, stream>>>(bufA, src, dst, ex, denom, bufB);
    add_bias_kernel<<<ncBlocks, TB, 0, stream>>>(bufB, b2, NN * CC);

    // ---------------- Edge predictor ----------------
    pred_init_kernel<<<(EE + TB - 1) / TB, TB, 0, stream>>>(out, bp2, EE);
    pred_gemm_kernel<<<dim3(EE / 128, CC / 64), gemmBlk, 0, stream>>>(
        bufB, src, dst, Wp1, bp1, wp2, out);
}
